// LocalAttention_40175124087389
// MI455X (gfx1250) — matrix operand order repaired and checked by reading, not compiled, not run on hardware
//
#include <hip/hip_runtime.h>
#include <hip/hip_bf16.h>

// Problem constants (reference: B=2, T=8192, C=1024, H=16, HD=64, WIN=256; pad_end==0)
#define BN    2
#define TN    8192
#define CN    1024
#define HN    16
#define HD    64
#define WIN_  256
#define HALFW 128
#define NW    32      // windows per batch (T/WIN)
#define CTX   512     // 2*WIN context keys per window
#define KSTEPS (CN / 32)

typedef _Float16 h16 __attribute__((ext_vector_type(16)));
typedef _Float16 h8  __attribute__((ext_vector_type(8)));
typedef _Float16 h4  __attribute__((ext_vector_type(4)));
typedef float    f8  __attribute__((ext_vector_type(8)));
typedef float    f4  __attribute__((ext_vector_type(4)));

union HU { h16 v; h8 h[2]; _Float16 e[16]; };

// Load 16 contiguous halves (two 16B loads) -> B fragment when source is K-contiguous per lane.
static __device__ __forceinline__ h16 ld_h16(const _Float16* p) {
  HU u; u.h[0] = *(const h8*)(p); u.h[1] = *(const h8*)(p + 8); return u.v;
}
// A fragment 16x32 f16 from row-major [m][k]: lane m=l&15, khalf=l>>4;
// halves 0..15 -> K = kh*16..kh*16+15 (contiguous, matches B fragments)
static __device__ __forceinline__ h16 ld_afrag(const _Float16* row_base, int ld, int lane, int k0) {
  const int m = lane & 15, kh = lane >> 4;
  const _Float16* p = row_base + (size_t)m * ld + k0 + kh * 16;
  HU u; u.h[0] = *(const h8*)(p); u.h[1] = *(const h8*)(p + 8); return u.v;
}
static __device__ __forceinline__ h16 h16_splat(float x) {
  HU u;
#pragma unroll
  for (int i = 0; i < 16; ++i) u.e[i] = (_Float16)x;
  return u.v;
}
static __device__ __forceinline__ f8 wmma16(h16 a, h16 b, f8 c) {
  return __builtin_amdgcn_wmma_f32_16x16x32_f16(false, a, false, b, (short)0, c, false, false);
}

// ---------------- pack kernels ----------------

// Pack Wq,Wk,Wv,Wo (each [cout][cin] fp32) into f16 B-fragment-native layout:
// Wp[z][ntile(64)][ktile(32)][lane(32)][h(16)], value = W[cout=ntile*16+(lane&15)][cin=ktile*32+(lane>>4)*16+h]
__global__ __launch_bounds__(256) void pack_w_kernel(
    const float* __restrict__ Wq, const float* __restrict__ Wk,
    const float* __restrict__ Wv, const float* __restrict__ Wo,
    _Float16* __restrict__ Wp)
{
  size_t idx = (size_t)blockIdx.x * 256 + threadIdx.x;   // 4*1024*1024 total
  const int h = (int)(idx & 15);
  size_t r = idx >> 4;
  const int lane = (int)(r & 31); r >>= 5;
  const int kt   = (int)(r & 31); r >>= 5;
  const int nt   = (int)(r & 63); r >>= 6;
  const int z    = (int)r;
  const int cout = nt * 16 + (lane & 15);
  const int cin  = kt * 32 + (lane >> 4) * 16 + h;
  const float* W = (z == 0) ? Wq : (z == 1) ? Wk : (z == 2) ? Wv : Wo;
  Wp[idx] = (_Float16)W[(size_t)cout * CN + cin];
}

__global__ __launch_bounds__(256) void pack_x_kernel(const float* __restrict__ x,
                                                     _Float16* __restrict__ xh)
{
  size_t i = ((size_t)blockIdx.x * 256 + threadIdx.x) * 4;
  f4 v = *(const f4*)(x + i);
  h4 o;
#pragma unroll
  for (int j = 0; j < 4; ++j) o[j] = (_Float16)v[j];
  *(h4*)(xh + i) = o;
}

// B-fragment load from packed weights at (ntile-of-head, kstep)
static __device__ __forceinline__ h16 ld_wfrag(const _Float16* __restrict__ Wb, int nt, int kt, int lane) {
  return ld_h16(Wb + (((size_t)nt * 32 + kt) * 32 + lane) * 16);
}

// ---------------- fused QKV GEMM + bias + RoPE ----------------
// grid = (16384/128, 16 heads, 3{q,k,v}); block = 256 (8 waves); wave = one 16-row strip x full head.
// Ping-pong double buffering (unroll-by-2): buffers filled directly by loads, no register copies.
__global__ __launch_bounds__(256) void qkv_kernel(
    const _Float16* __restrict__ xh, const _Float16* __restrict__ Wp,
    const float* __restrict__ bqp, const float* __restrict__ bkp, const float* __restrict__ bvp,
    const float* __restrict__ cosT, const float* __restrict__ sinT,
    _Float16* __restrict__ qh, _Float16* __restrict__ kh, _Float16* __restrict__ vt)
{
  const int lane = threadIdx.x & 31;
  const int wave = threadIdx.x >> 5;
  const int z    = blockIdx.z;
  const int head = blockIdx.y;
  const int mrow = blockIdx.x * 128 + wave * 16;   // token index in [0, B*T)
  const int mL   = lane & 15;
  const _Float16* A  = xh + (size_t)mrow * CN;
  const _Float16* Wb = Wp + (size_t)z * CN * CN + ((size_t)(head * 4) * 32) * 32 * 16;

  f8 acc[4] = {};
  h16 a0 = ld_afrag(A, CN, lane, 0);
  h16 b0[4];
#pragma unroll
  for (int nt = 0; nt < 4; ++nt) b0[nt] = ld_wfrag(Wb, nt, 0, lane);

  for (int kt = 0; kt < KSTEPS; kt += 2) {
    h16 a1 = ld_afrag(A, CN, lane, (kt + 1) * 32);
    h16 b1[4];
#pragma unroll
    for (int nt = 0; nt < 4; ++nt) b1[nt] = ld_wfrag(Wb, nt, kt + 1, lane);
#pragma unroll
    for (int nt = 0; nt < 4; ++nt) acc[nt] = wmma16(a0, b0[nt], acc[nt]);
    if (kt + 2 < KSTEPS) {
      a0 = ld_afrag(A, CN, lane, (kt + 2) * 32);
#pragma unroll
      for (int nt = 0; nt < 4; ++nt) b0[nt] = ld_wfrag(Wb, nt, kt + 2, lane);
      __builtin_prefetch(A + (size_t)mL * CN + (kt + 3) * 32, 0, 3);
    }
#pragma unroll
    for (int nt = 0; nt < 4; ++nt) acc[nt] = wmma16(a1, b1[nt], acc[nt]);
  }

  const int bI = mrow >> 13;           // batch (T=8192)
  const int t0 = mrow & (TN - 1);
  const float* bias = (z == 0) ? bqp : (z == 1) ? bkp : bvp;
  float vv[4][8];
#pragma unroll
  for (int nt = 0; nt < 4; ++nt) {
    const float bb = bias[head * HD + nt * 16 + mL];
#pragma unroll
    for (int r = 0; r < 8; ++r) vv[nt][r] = acc[nt][r] + bb;
  }
  if (z < 2) { // RoPE on q,k: pairs (d, d+-32) live in tiles nt and nt+-2 (same lane)
    float og[4][8];
#pragma unroll
    for (int nt = 0; nt < 4; ++nt)
#pragma unroll
      for (int r = 0; r < 8; ++r) og[nt][r] = vv[nt][r];
#pragma unroll
    for (int r = 0; r < 8; ++r) {
      const int t = t0 + ((lane & 16) ? 8 : 0) + r;   // D-tile row: lanes<16 -> M=r, lanes>=16 -> M=8+r
#pragma unroll
      for (int nt = 0; nt < 4; ++nt) {
        const int dh = nt * 16 + mL;
        const float c = cosT[t * HD + dh];
        const float s = sinT[t * HD + dh];
        const float rot = (nt < 2) ? -og[nt + 2][r] : og[nt - 2][r];
        vv[nt][r] = og[nt][r] * c + rot * s;
      }
    }
  }
  if (z == 2) { // V stored transposed [b][h][hd][t] -> contiguous B-frags for P@V
    _Float16* dst = vt + (size_t)(bI * HN + head) * HD * TN;
#pragma unroll
    for (int r = 0; r < 8; ++r) {
      const int t = t0 + ((lane & 16) ? 8 : 0) + r;
#pragma unroll
      for (int nt = 0; nt < 4; ++nt)
        dst[(size_t)(nt * 16 + mL) * TN + t] = (_Float16)vv[nt][r];
    }
  } else {      // Q,K row-major [b][h][t][hd]
    _Float16* dst = ((z == 0) ? qh : kh) + (size_t)(bI * HN + head) * TN * HD;
#pragma unroll
    for (int r = 0; r < 8; ++r) {
      const int t = t0 + ((lane & 16) ? 8 : 0) + r;
#pragma unroll
      for (int nt = 0; nt < 4; ++nt)
        dst[(size_t)t * HD + nt * 16 + mL] = (_Float16)vv[nt][r];
    }
  }
}

// ---------------- attend-mask build (incl. all_attend quirk) ----------------
__global__ __launch_bounds__(512) void mask_kernel(const unsigned char* __restrict__ pm,
                                                   unsigned char* __restrict__ msk)
{
  const int w = blockIdx.x;          // 0..63 global window
  const int j = threadIdx.x;         // 0..511
  const int b = w >> 5, wl = w & 31;
  const int p = wl * WIN_ - HALFW + j;
  int attend = 1;                    // halo positions: pk padded False -> attend
  if (p >= 0 && p < TN) attend = pm[b * TN + p] ? 0 : 1;
  __shared__ int allatt;
  if (j == 0) allatt = 1;
  __syncthreads();
  if (!attend) atomicAnd(&allatt, 0);
  __syncthreads();
  if (allatt && j == 0) attend = 0;  // reference: mask key 0 when everything attendable
  msk[w * CTX + j] = (unsigned char)attend;
}

// ---------------- attention fragment loaders ----------------
static __device__ __forceinline__ void load_kfrags(const _Float16* __restrict__ K, int wl, int j0,
                                                   int mL, int khf, h16 kb[2][4])
{
#pragma unroll
  for (int kk = 0; kk < 2; ++kk)
#pragma unroll
    for (int nt = 0; nt < 4; ++nt) {
      const int p = wl * WIN_ - HALFW + j0 + nt * 16 + mL;  // key position (column of B-frag)
      kb[kk][nt] = (p >= 0 && p < TN) ? ld_h16(K + (size_t)p * HD + kk * 32 + khf * 16)
                                      : h16_splat(0.0f);    // halo keys are 0 after padded RoPE
    }
}
static __device__ __forceinline__ void load_vfrags(const _Float16* __restrict__ V,
                                                   const float* __restrict__ bvp, int head,
                                                   int wl, int j0, int mL, int khf, h16 vb[2][4])
{
#pragma unroll
  for (int kk = 0; kk < 2; ++kk) {
    const int p0 = wl * WIN_ - HALFW + j0 + kk * 32 + khf * 16;  // 16 contiguous keys per lane
#pragma unroll
    for (int nt = 0; nt < 4; ++nt) {
      const int dh = nt * 16 + mL;
      vb[kk][nt] = (p0 >= 0 && p0 < TN) ? ld_h16(V + (size_t)dh * TN + p0)
                                        : h16_splat(bvp[head * HD + dh]); // halo value rows = bv
    }
  }
}

// ---------------- one 64-key flash-attention chunk ----------------
static __device__ __forceinline__ void attn_chunk(
    int j0, int wl, int w, int head, int mL, int khf,
    const h16 qa[2][2], const h16 kb[2][4],
    const _Float16* __restrict__ V, const float* __restrict__ bvp,
    const unsigned char* __restrict__ msk,
    _Float16 (* __restrict__ ldsW)[72],       // this wave's [32][72] LDS slab
    float mrun[2][8], float lrun[2][8], f8 oacc[2][4])
{
  // ---- issue V loads early: consumed only after softmax ----
  h16 vb[2][4];
  load_vfrags(V, bvp, head, wl, j0, mL, khf, vb);
  // ---- S = Q * K^T ----
  f8 s[2][4] = {};
#pragma unroll
  for (int kk = 0; kk < 2; ++kk)
#pragma unroll
    for (int nt = 0; nt < 4; ++nt)
#pragma unroll
      for (int mt = 0; mt < 2; ++mt)
        s[mt][nt] = wmma16(qa[mt][kk], kb[kk][nt], s[mt][nt]);
  // ---- scale + mask ----
#pragma unroll
  for (int nt = 0; nt < 4; ++nt) {
    const unsigned char att = msk[w * CTX + j0 + nt * 16 + mL];
#pragma unroll
    for (int mt = 0; mt < 2; ++mt)
#pragma unroll
      for (int r = 0; r < 8; ++r) {
        const float val = s[mt][nt][r] * 0.125f;            // 1/sqrt(64)
        s[mt][nt][r] = att ? val : -3.0e38f;
      }
  }
  // ---- online softmax per row (row values live on 16 lanes of one half) ----
#pragma unroll
  for (int mt = 0; mt < 2; ++mt)
#pragma unroll
    for (int r = 0; r < 8; ++r) {
      float cm = fmaxf(fmaxf(s[mt][0][r], s[mt][1][r]), fmaxf(s[mt][2][r], s[mt][3][r]));
#pragma unroll
      for (int d = 1; d <= 8; d <<= 1) cm = fmaxf(cm, __shfl_xor(cm, d));
      const float mnew = fmaxf(mrun[mt][r], cm);
      const float corr = __expf(mrun[mt][r] - mnew);
      mrun[mt][r] = mnew;
      float rs = 0.0f;
#pragma unroll
      for (int nt = 0; nt < 4; ++nt) {
        const float pv = __expf(s[mt][nt][r] - mnew);
        s[mt][nt][r] = pv;
        rs += pv;
      }
#pragma unroll
      for (int d = 1; d <= 8; d <<= 1) rs += __shfl_xor(rs, d);
      lrun[mt][r] = lrun[mt][r] * corr + rs;
#pragma unroll
      for (int nt = 0; nt < 4; ++nt) oacc[mt][nt][r] *= corr;
    }
  // ---- stage P through LDS: D-tile layout -> row-major [qrow][key] ----
#pragma unroll
  for (int mt = 0; mt < 2; ++mt)
#pragma unroll
    for (int nt = 0; nt < 4; ++nt)
#pragma unroll
      for (int r = 0; r < 8; ++r) {
        const int row = mt * 16 + ((mL != (threadIdx.x & 31)) ? 0 : 0) + ((threadIdx.x & 16) ? 8 : 0) + r;
        ldsW[row][nt * 16 + mL] = (_Float16)s[mt][nt][r];
      }
  asm volatile("s_wait_dscnt 0" ::: "memory");  // per-wave LDS store->load ordering
  // ---- O += P * V ----
#pragma unroll
  for (int kk = 0; kk < 2; ++kk) {
    h16 pa[2];
#pragma unroll
    for (int mt = 0; mt < 2; ++mt) {
      const _Float16* p = &ldsW[mt * 16 + mL][kk * 32 + khf * 16];
      HU u; u.h[0] = *(const h8*)(p); u.h[1] = *(const h8*)(p + 8);
      pa[mt] = u.v;
    }
#pragma unroll
    for (int nt = 0; nt < 4; ++nt)
#pragma unroll
      for (int mt = 0; mt < 2; ++mt)
        oacc[mt][nt] = wmma16(pa[mt], vb[kk][nt], oacc[mt][nt]);
  }
}

// ---------------- fused flash attention per (window, head) ----------------
// block = 256 (8 waves); wave = 32 query rows; context streamed as 8 chunks of 64 keys,
// two chunks per loop iteration with ping-pong K fragment buffers (no register copies).
__global__ __launch_bounds__(256) void attn_kernel(
    const _Float16* __restrict__ qh, const _Float16* __restrict__ kh, const _Float16* __restrict__ vt,
    const float* __restrict__ bvp, const unsigned char* __restrict__ msk, _Float16* __restrict__ ao)
{
  const int lane = threadIdx.x & 31;
  const int wave = threadIdx.x >> 5;
  const int wl   = blockIdx.x;       // window in batch
  const int head = blockIdx.y;
  const int b    = blockIdx.z;
  const int w    = b * NW + wl;
  const int mL   = lane & 15;
  const int khf  = lane >> 4;
  const _Float16* Q = qh + (size_t)(b * HN + head) * TN * HD;
  const _Float16* K = kh + (size_t)(b * HN + head) * TN * HD;
  const _Float16* V = vt + (size_t)(b * HN + head) * HD * TN;
  const int q0 = wl * WIN_ + wave * 32;

  __shared__ _Float16 ldsP[8][32][72];  // 72-half rows: 144B stride keeps h8 loads 16B aligned

  h16 qa[2][2];
#pragma unroll
  for (int mt = 0; mt < 2; ++mt)
#pragma unroll
    for (int kk = 0; kk < 2; ++kk)
      qa[mt][kk] = ld_afrag(Q + (size_t)(q0 + mt * 16) * HD, HD, lane, kk * 32);

  float mrun[2][8], lrun[2][8];
  f8 oacc[2][4] = {};
#pragma unroll
  for (int mt = 0; mt < 2; ++mt)
#pragma unroll
    for (int r = 0; r < 8; ++r) { mrun[mt][r] = -3.0e38f; lrun[mt][r] = 0.0f; }

  h16 kb0[2][4], kb1[2][4];
  load_kfrags(K, wl, 0, mL, khf, kb0);

  for (int j0 = 0; j0 < CTX; j0 += 128) {
    load_kfrags(K, wl, j0 + 64, mL, khf, kb1);
    attn_chunk(j0, wl, w, head, mL, khf, qa, kb0, V, bvp, msk, ldsP[wave], mrun, lrun, oacc);
    if (j0 + 128 < CTX) load_kfrags(K, wl, j0 + 128, mL, khf, kb0);
    attn_chunk(j0 + 64, wl, w, head, mL, khf, qa, kb1, V, bvp, msk, ldsP[wave], mrun, lrun, oacc);
  }

  // ---- normalize + store attention output f16 [token][C] for O-projection ----
#pragma unroll
  for (int mt = 0; mt < 2; ++mt)
#pragma unroll
    for (int r = 0; r < 8; ++r) {
      const float inv = 1.0f / lrun[mt][r];
      const int t = q0 + mt * 16 + ((lane & 16) ? 8 : 0) + r;
#pragma unroll
      for (int nt = 0; nt < 4; ++nt)
        ao[(size_t)(b * TN + t) * CN + head * HD + nt * 16 + mL] =
            (_Float16)(oacc[mt][nt][r] * inv);
    }
}

// ---------------- output projection GEMM (fp32 out), ping-pong double-buffered ----------------
__global__ __launch_bounds__(256) void oproj_kernel(
    const _Float16* __restrict__ ao, const _Float16* __restrict__ Wpo,
    const float* __restrict__ bop, float* __restrict__ out)
{
  const int lane = threadIdx.x & 31;
  const int wave = threadIdx.x >> 5;
  const int mrow = blockIdx.x * 128 + wave * 16;
  const int nb   = blockIdx.y;          // cout block of 64
  const int mL   = lane & 15;
  const _Float16* A  = ao + (size_t)mrow * CN;
  const _Float16* Wb = Wpo + ((size_t)(nb * 4) * 32) * 32 * 16;

  f8 acc[4] = {};
  h16 a0 = ld_afrag(A, CN, lane, 0);
  h16 b0[4];
#pragma unroll
  for (int nt = 0; nt < 4; ++nt) b0[nt] = ld_wfrag(Wb, nt, 0, lane);

  for (int kt = 0; kt < KSTEPS; kt += 2) {
    h16 a1 = ld_afrag(A, CN, lane, (kt + 1) * 32);
    h16 b1[4];
#pragma unroll
    for (int nt = 0; nt < 4; ++nt) b1[nt] = ld_wfrag(Wb, nt, kt + 1, lane);
#pragma unroll
    for (int nt = 0; nt < 4; ++nt) acc[nt] = wmma16(a0, b0[nt], acc[nt]);
    if (kt + 2 < KSTEPS) {
      a0 = ld_afrag(A, CN, lane, (kt + 2) * 32);
#pragma unroll
      for (int nt = 0; nt < 4; ++nt) b0[nt] = ld_wfrag(Wb, nt, kt + 2, lane);
      __builtin_prefetch(A + (size_t)mL * CN + (kt + 3) * 32, 0, 3);
    }
#pragma unroll
    for (int nt = 0; nt < 4; ++nt) acc[nt] = wmma16(a1, b1[nt], acc[nt]);
  }

#pragma unroll
  for (int nt = 0; nt < 4; ++nt) {
    const int c = nb * 64 + nt * 16 + mL;
    const float bb = bop[c];
#pragma unroll
    for (int r = 0; r < 8; ++r) {
      const int t = mrow + ((lane & 16) ? 8 : 0) + r;
      out[(size_t)t * CN + c] = acc[nt][r] + bb;
    }
  }
}

// ---------------- host launcher ----------------
extern "C" void kernel_launch(void* const* d_in, const int* in_sizes, int n_in,
                              void* d_out, int out_size, void* d_ws, size_t ws_size,
                              hipStream_t stream) {
  (void)in_sizes; (void)n_in; (void)out_size; (void)ws_size;
  const float*         x    = (const float*)d_in[0];
  const unsigned char* pm   = (const unsigned char*)d_in[1];   // jax bool -> 1 byte
  const float*         cosT = (const float*)d_in[2];           // [T,64]
  const float*         sinT = (const float*)d_in[3];
  const float* Wq = (const float*)d_in[4];  const float* bq = (const float*)d_in[5];
  const float* Wk = (const float*)d_in[6];  const float* bk = (const float*)d_in[7];
  const float* Wv = (const float*)d_in[8];  const float* bv = (const float*)d_in[9];
  const float* Wo = (const float*)d_in[10]; const float* bo = (const float*)d_in[11];
  float* out = (float*)d_out;

  // Workspace layout (needs 136 MB + 32 KB):
  char* ws = (char*)d_ws;
  _Float16* Wp = (_Float16*)ws;                                      //  8 MB: packed Wq,Wk,Wv,Wo
  _Float16* xh = (_Float16*)(ws + (size_t)8   * 1024 * 1024);        // 32 MB: x f16 (reused as attn out)
  _Float16* qh = (_Float16*)(ws + (size_t)40  * 1024 * 1024);        // 32 MB: Q roped [b][h][t][hd]
  _Float16* kk = (_Float16*)(ws + (size_t)72  * 1024 * 1024);        // 32 MB: K roped [b][h][t][hd]
  _Float16* vt = (_Float16*)(ws + (size_t)104 * 1024 * 1024);        // 32 MB: V^T   [b][h][hd][t]
  unsigned char* msk = (unsigned char*)(ws + (size_t)136 * 1024 * 1024); // 32 KB
  _Float16* ao = xh;  // attention output reuses xh (xh dead after qkv_kernel)

  pack_w_kernel<<<16384, 256, 0, stream>>>(Wq, Wk, Wv, Wo, Wp);
  pack_x_kernel<<<16384, 256, 0, stream>>>(x, xh);                   // 16M elems, 4/thread
  qkv_kernel<<<dim3(128, 16, 3), 256, 0, stream>>>(xh, Wp, bq, bk, bv, cosT, sinT, qh, kk, vt);
  mask_kernel<<<64, 512, 0, stream>>>(pm, msk);
  attn_kernel<<<dim3(32, 16, 2), 256, 0, stream>>>(qh, kk, vt, bv, msk, ao);
  oproj_kernel<<<dim3(128, 16), 256, 0, stream>>>(ao, Wp + (size_t)3 * CN * CN, bo, out);
}
